// FeatureAdaption_2138893713705
// MI455X (gfx1250) — matrix operand order repaired and checked by reading, not compiled, not run on hardware
//
#include <hip/hip_runtime.h>
#include <hip/hip_bf16.h>

// Problem constants (from reference)
#define BB    4
#define CIN   256
#define COUT  256
#define HH    64
#define WW    64
#define KKQ   9          // K*K
#define GG    4          // deform groups
#define HW    4096       // H*W
#define KDIM  2304       // CIN*KK
#define NPIX  16384      // B*H*W

typedef __attribute__((ext_vector_type(16))) __bf16 bf16x16;
typedef __attribute__((ext_vector_type(8)))  float  v8f;

union Frag { bf16x16 v; uint4 q[2]; };

__device__ __forceinline__ unsigned int f2bf(float f) {
  unsigned int u = __float_as_uint(f);
  return (u + 0x7FFFu + ((u >> 16) & 1u)) >> 16;   // round-to-nearest-even
}

// ---------------------------------------------------------------------------
// Kernel 1: convert w_deform (COUT,CIN,3,3) f32 -> bf16, layout [o][c*9+kk]
// ---------------------------------------------------------------------------
__global__ void cvt_w_kernel(const float* __restrict__ w,
                             unsigned int* __restrict__ wb) {
  int i = blockIdx.x * blockDim.x + threadIdx.x;     // pair index, 294912 total
  float a = w[2 * i], b = w[2 * i + 1];
  wb[i] = f2bf(a) | (f2bf(b) << 16);
}

// ---------------------------------------------------------------------------
// Kernel 2: fused offset-conv + bilinear sampling (im2col in bf16)
// sampled[n][k], n = b*4096 + h*64 + w, k = c*9 + kk. One wave per pixel.
// ---------------------------------------------------------------------------
__device__ __forceinline__ float sample1(const float* __restrict__ ch,
                                         int yi, int xi) {
  int cy = yi < 0 ? 0 : (yi > 63 ? 63 : yi);
  int cx = xi < 0 ? 0 : (xi > 63 ? 63 : xi);
  float v = ch[(cy << 6) + cx];
  bool ok = (yi >= 0) & (yi < 64) & (xi >= 0) & (xi < 64);
  return ok ? v : 0.0f;
}

__global__ void __launch_bounds__(256)
sample_kernel(const float* __restrict__ x, const float* __restrict__ y,
              const float* __restrict__ woff, unsigned int* __restrict__ samp) {
  __shared__ float coords[8][40][2];                 // 36 used per wave
  const int wave = threadIdx.x >> 5;
  const int lane = threadIdx.x & 31;
  const int n = (blockIdx.x << 3) + wave;            // pixel index
  const int b = n >> 12, p = n & 4095;
  const int h = p >> 6,  w = p & 63;

  const float* yb = y + (size_t)b * 2 * HW + p;
  const float y0v = yb[0], y1v = yb[HW];

  // 36 (g,kk) deformed coordinates for this pixel (fused offset 1x1 conv)
  for (int i = lane; i < GG * KKQ; i += 32) {
    int g  = i / 9, kk = i - 9 * g;
    int ky = kk / 3, kx = kk - 3 * ky;
    const float* wo = woff + i * 4;                  // rows 2i (dy), 2i+1 (dx)
    float dy = wo[0] * y0v + wo[1] * y1v;
    float dx = wo[2] * y0v + wo[3] * y1v;
    coords[wave][i][0] = dy + (float)(h - 1 + ky);
    coords[wave][i][1] = dx + (float)(w - 1 + kx);
    (void)g;
  }
  __syncthreads();

  const float* xb = x + (size_t)b * CIN * HW;
  unsigned int* row = samp + (size_t)n * (KDIM / 2); // 2 bf16 per dword
  for (int t = 0; t < 36; ++t) {
    unsigned int pack = 0;
#pragma unroll
    for (int j = 0; j < 2; ++j) {
      int k  = t * 64 + lane * 2 + j;
      int c  = k / 9;
      int kk = k - 9 * c;
      int g  = c >> 6;
      float yc = coords[wave][g * 9 + kk][0];
      float xc = coords[wave][g * 9 + kk][1];
      float y0f = floorf(yc), x0f = floorf(xc);
      float wy = yc - y0f,  wx = xc - x0f;
      int yi = (int)y0f,    xi = (int)x0f;
      const float* ch = xb + c * HW;
      float v00 = sample1(ch, yi,     xi);
      float v01 = sample1(ch, yi,     xi + 1);
      float v10 = sample1(ch, yi + 1, xi);
      float v11 = sample1(ch, yi + 1, xi + 1);
      float v = v00 * (1.f - wy) * (1.f - wx) + v01 * (1.f - wy) * wx
              + v10 * wy * (1.f - wx)         + v11 * wy * wx;
      pack |= f2bf(v) << (16 * j);
    }
    row[t * 32 + lane] = pack;                       // coalesced 128B/wave
  }
}

// ---------------------------------------------------------------------------
// Kernel 3: WMMA bf16 GEMM  out[b,o,hw] = relu( W[256x2304] * S^T[2304x16384] )
// Block tile 128x128, KT=32, 8 waves -> each wave 64x32 (4x2 WMMA frags).
// ---------------------------------------------------------------------------
#define MT  128
#define NT  128
#define KT  32
#define LDP 40   // padded LDS row stride (bf16 elems) -> 20-dword stride

__global__ void __launch_bounds__(256, 1)
gemm_kernel(const __bf16* __restrict__ wb, const __bf16* __restrict__ samp,
            float* __restrict__ out) {
  __shared__ __bf16 As[MT][LDP];
  __shared__ __bf16 Bs[NT][LDP];

  const int tid    = threadIdx.x;
  const int lane   = tid & 31;
  const int waveId = tid >> 5;
  const int wm = waveId & 1;           // 0..1 -> M offset wm*64
  const int wn = waveId >> 1;          // 0..3 -> N offset wn*32
  const int mBase = blockIdx.y * MT;
  const int nBase = blockIdx.x * NT;

  // Global->LDS staging: thread t covers rows (t>>2) and (t>>2)+64, 16B seg
  const int r0  = tid >> 2;            // 0..63
  const int seg = (tid & 3) * 8;       // bf16 element offset in row
  const __bf16* pa0 = wb   + (size_t)(mBase + r0) * KDIM + seg;
  const __bf16* pa1 = pa0  + (size_t)64 * KDIM;
  const __bf16* pb0 = samp + (size_t)(nBase + r0) * KDIM + seg;
  const __bf16* pb1 = pb0  + (size_t)64 * KDIM;

  uint4 ra0 = *(const uint4*)pa0;
  uint4 ra1 = *(const uint4*)pa1;
  uint4 rb0 = *(const uint4*)pb0;
  uint4 rb1 = *(const uint4*)pb1;

  v8f acc[4][2];
#pragma unroll
  for (int i = 0; i < 4; ++i)
#pragma unroll
    for (int j = 0; j < 2; ++j)
#pragma unroll
      for (int r = 0; r < 8; ++r) acc[i][j][r] = 0.0f;

  const int half = lane >> 4;          // ISA 7.12.2 lane-half split
  const int lrow = lane & 15;
  const int kbA  = half * 8;           // A: K {0..7,16..23} / {8..15,24..31}
  const int kbB  = half * 8;           // B: K {0..7,16..23} / {8..15,24..31}

  const int NKT = KDIM / KT;           // 72
  for (int kt = 0; kt < NKT; ++kt) {
    __syncthreads();
    *(uint4*)&As[r0][seg]      = ra0;
    *(uint4*)&As[r0 + 64][seg] = ra1;
    *(uint4*)&Bs[r0][seg]      = rb0;
    *(uint4*)&Bs[r0 + 64][seg] = rb1;
    __syncthreads();

    if (kt + 1 < NKT) {                // register prefetch of next K-tile
      int off = (kt + 1) * KT;
      ra0 = *(const uint4*)(pa0 + off);
      ra1 = *(const uint4*)(pa1 + off);
      rb0 = *(const uint4*)(pb0 + off);
      rb1 = *(const uint4*)(pb1 + off);
      if (kt + 2 < NKT) {              // L2 prefetch of the tile after next
        __builtin_prefetch(pa0 + off + KT, 0, 0);
        __builtin_prefetch(pb0 + off + KT, 0, 0);
      }
    }

    Frag a[4], fb[2];
#pragma unroll
    for (int i = 0; i < 4; ++i) {
      const __bf16* p = &As[wm * 64 + i * 16 + lrow][kbA];
      a[i].q[0] = *(const uint4*)p;
      a[i].q[1] = *(const uint4*)(p + 16);
    }
#pragma unroll
    for (int j = 0; j < 2; ++j) {
      const __bf16* p = &Bs[wn * 32 + j * 16 + lrow][kbB];
      fb[j].q[0] = *(const uint4*)p;
      fb[j].q[1] = *(const uint4*)(p + 16);
    }
#pragma unroll
    for (int i = 0; i < 4; ++i)
#pragma unroll
      for (int j = 0; j < 2; ++j)
        acc[i][j] = __builtin_amdgcn_wmma_f32_16x16x32_bf16(
            false, a[i].v, false, fb[j].v, (short)0, acc[i][j], false, false);
  }

  // Epilogue: ReLU + store. C/D layout: lanes 0-15 M=r, lanes 16-31 M=r+8.
#pragma unroll
  for (int i = 0; i < 4; ++i) {
#pragma unroll
    for (int j = 0; j < 2; ++j) {
      int m  = mBase + wm * 64 + i * 16 + half * 8;
      int nG = nBase + wn * 32 + j * 16 + lrow;
      int b  = nG >> 12, p = nG & 4095;           // NT=128 never straddles batch
      float* op = out + ((size_t)((b << 8) + m) << 12) + p;
#pragma unroll
      for (int r = 0; r < 8; ++r) {
        float v = acc[i][j][r];
        op[(size_t)r * HW] = v > 0.f ? v : 0.f;
      }
    }
  }
}

// ---------------------------------------------------------------------------
extern "C" void kernel_launch(void* const* d_in, const int* in_sizes, int n_in,
                              void* d_out, int out_size, void* d_ws, size_t ws_size,
                              hipStream_t stream) {
  const float* x        = (const float*)d_in[0];   // (4,256,64,64)
  const float* y        = (const float*)d_in[1];   // (4,2,64,64)
  const float* w_offset = (const float*)d_in[2];   // (72,2,1,1)
  const float* w_deform = (const float*)d_in[3];   // (256,256,3,3)
  float* out = (float*)d_out;                      // (4,256,64,64)

  // Workspace: [sampled bf16 NPIX*KDIM][weights bf16 COUT*KDIM]
  const size_t sampBytes = (size_t)NPIX * KDIM * sizeof(__bf16); // 75.5 MB (fits L2)
  __bf16* samp = (__bf16*)d_ws;
  __bf16* wbf  = (__bf16*)((char*)d_ws + sampBytes);

  // 1) weights f32 -> bf16 (294912 dword pairs)
  cvt_w_kernel<<<(COUT * KDIM / 2) / 256, 256, 0, stream>>>(
      w_deform, (unsigned int*)wbf);

  // 2) fused offset conv + bilinear im2col (one wave per pixel, 8 waves/block)
  sample_kernel<<<NPIX / 8, 256, 0, stream>>>(
      x, y, w_offset, (unsigned int*)samp);

  // 3) WMMA GEMM + ReLU: grid = (N/128, M/128) = (128, 2)
  gemm_kernel<<<dim3(NPIX / NT, COUT / MT), 256, 0, stream>>>(wbf, samp, out);
}
